// GraphCut_43190191128913
// MI455X (gfx1250) — matrix operand order repaired and checked by reading, not compiled, not run on hardware
//
#include <hip/hip_runtime.h>

typedef __attribute__((ext_vector_type(16))) _Float16 v16h;
typedef __attribute__((ext_vector_type(8)))  _Float16 v8h;
typedef __attribute__((ext_vector_type(8)))  float    v8f;

#define NTOT 8192      // V*B rows of cf
#define BATCH 4096
#define DIM 128
#define NSEG 2         // column split factor
#define COLS_PER_SEG (NTOT / NSEG)
#define LOG2E 1.4426950408889634f
#define LN2   0.6931471805599453f
#define C2 (LOG2E / 0.7f)          // sim -> log2 domain scale (1/T folded in)
#define LOSS_SCALE 10.0f           // TEMPERATURE / BASE_TEMPERATURE
#define LOG2_LAMDA (-1.0f)         // log2(0.5)

#if __has_builtin(__builtin_amdgcn_exp2f)
#define EXP2F(x) __builtin_amdgcn_exp2f(x)
#else
#define EXP2F(x) exp2f(x)
#endif
#if __has_builtin(__builtin_amdgcn_logf)
#define LOG2F(x) __builtin_amdgcn_logf(x)
#else
#define LOG2F(x) log2f(x)
#endif

// ---------------------------------------------------------------------------
// Kernel 1: pack cf (view-major) into split-f16 hi/lo panels + widened labels.
// cf[v*BATCH + b][d] = features[b][v][d];  hi = f16(x), lo = f16(x - hi)
// ---------------------------------------------------------------------------
__global__ void __launch_bounds__(256)
supcon_pack(const float* __restrict__ feat, const int* __restrict__ labels,
            _Float16* __restrict__ cf_hi, _Float16* __restrict__ cf_lo,
            int* __restrict__ lab) {
  int i = blockIdx.x * blockDim.x + threadIdx.x;
  if (i < NTOT * DIM) {
    int row = i >> 7;            // i / DIM
    int d   = i & (DIM - 1);
    int v   = row >> 12;         // row / BATCH
    int b   = row & (BATCH - 1);
    float x = feat[(size_t)b * (2 * DIM) + v * DIM + d];
    _Float16 hi = (_Float16)x;
    _Float16 lo = (_Float16)(x - (float)hi);
    cf_hi[i] = hi;
    cf_lo[i] = lo;
  }
  if (i < NTOT) lab[i] = labels[i & (BATCH - 1)];
}

// ---------------------------------------------------------------------------
// Kernel 2: fused sim-GEMM (split-f16 WMMA) + online masked softmax partials.
// One wave32 per (16-row tile, column segment). 12 v_wmma_f32_16x16x32_f16
// per 16x16 tile: hi*hi + hi*lo + lo*hi (lo*lo ~2^-22, dropped).
//
// Layouts used here (16-bit, contiguous k; not the ISA table order):
//   A 16x32: lane m=L&15; elems 0-15 = K k0+16*half (consecutive)
//   B 32x16: lane n=L&15; elems 0-15 = K k0+16*half (consecutive)
//   C 16x16 f32: vgpr r: lanes 0-15 -> M=r, N=lane; lanes 16-31 -> M=r+8
// ---------------------------------------------------------------------------
__global__ void __launch_bounds__(32)
supcon_main(const _Float16* __restrict__ cf_hi, const _Float16* __restrict__ cf_lo,
            const int* __restrict__ lab,
            float* __restrict__ pm, float* __restrict__ pp, float* __restrict__ pn) {
  const int lane = threadIdx.x & 31;
  const int lrow = lane & 15;
  const int half = lane >> 4;

  const int rt  = blockIdx.x >> 1;        // row tile 0..511
  const int seg = blockIdx.x & 1;         // column segment 0..1
  const int r0  = rt * 16;
  const int cbase = seg * COLS_PER_SEG;

  // ---- preload A panel (hi & lo) for K=128: 4 fragments each -------------
  v16h ah[4], al[4];
  {
    const _Float16* arh = cf_hi + (size_t)(r0 + lrow) * DIM;
    const _Float16* arl = cf_lo + (size_t)(r0 + lrow) * DIM;
#pragma unroll
    for (int k = 0; k < 4; ++k) {
      const int k0 = 32 * k;
      v8h h0 = *(const v8h*)(arh + k0 + 16 * half);
      v8h h1 = *(const v8h*)(arh + k0 + 16 * half + 8);
      v8h l0 = *(const v8h*)(arl + k0 + 16 * half);
      v8h l1 = *(const v8h*)(arl + k0 + 16 * half + 8);
      ah[k] = __builtin_shufflevector(h0, h1, 0,1,2,3,4,5,6,7,8,9,10,11,12,13,14,15);
      al[k] = __builtin_shufflevector(l0, l1, 0,1,2,3,4,5,6,7,8,9,10,11,12,13,14,15);
    }
  }

  int li[8];
#pragma unroll
  for (int r = 0; r < 8; ++r) li[r] = lab[r0 + r + 8 * half];

  // per-lane online softmax state in log2 domain
  float m[8], p[8], nn[8];
#pragma unroll
  for (int r = 0; r < 8; ++r) { m[r] = -3.0e38f; p[r] = 0.0f; nn[r] = 0.0f; }

  const _Float16* bbase_h = cf_hi + (size_t)lrow * DIM + 16 * half;
  const _Float16* bbase_l = cf_lo + (size_t)lrow * DIM + 16 * half;

  for (int ct = 0; ct < COLS_PER_SEG / 16; ++ct) {
    const int c0 = cbase + ct * 16;
    const int j  = c0 + lrow;
    const int lj = lab[j];
    const _Float16* bh_p = bbase_h + (size_t)c0 * DIM;
    const _Float16* bl_p = bbase_l + (size_t)c0 * DIM;

    v8f acc = {};
#pragma unroll
    for (int k = 0; k < 4; ++k) {
      v16h bh = *(const v16h*)(bh_p + 32 * k);
      v16h bl = *(const v16h*)(bl_p + 32 * k);
      acc = __builtin_amdgcn_wmma_f32_16x16x32_f16(false, ah[k], false, bh,
                                                   (short)0, acc, false, false);
      acc = __builtin_amdgcn_wmma_f32_16x16x32_f16(false, ah[k], false, bl,
                                                   (short)0, acc, false, false);
      acc = __builtin_amdgcn_wmma_f32_16x16x32_f16(false, al[k], false, bh,
                                                   (short)0, acc, false, false);
    }

    // online update (log2 domain): max includes diagonal, sums exclude it
#pragma unroll
    for (int r = 0; r < 8; ++r) {
      const int i = r0 + r + 8 * half;
      float x  = acc[r] * C2;
      float nm = fmaxf(m[r], x);
      float s  = EXP2F(m[r] - nm);
      float e  = (i == j) ? 0.0f : EXP2F(x - nm);
      float ep = (li[r] == lj) ? e : 0.0f;
      p[r]  = p[r]  * s + ep;
      nn[r] = nn[r] * s + (e - ep);
      m[r]  = nm;
    }
  }

  // merge (m,p,n) across the 16 lanes of each half; write row partials
#pragma unroll
  for (int r = 0; r < 8; ++r) {
#pragma unroll
    for (int d = 1; d < 16; d <<= 1) {
      float om = __shfl_xor(m[r],  d, 32);
      float op = __shfl_xor(p[r],  d, 32);
      float on = __shfl_xor(nn[r], d, 32);
      float nm = fmaxf(m[r], om);
      float s0 = EXP2F(m[r] - nm);
      float s1 = EXP2F(om   - nm);
      p[r]  = p[r]  * s0 + op * s1;
      nn[r] = nn[r] * s0 + on * s1;
      m[r]  = nm;
    }
    if (lrow == 0) {
      int idx = seg * NTOT + r0 + r + 8 * half;
      pm[idx] = m[r];
      pp[idx] = p[r];
      pn[idx] = nn[r];
    }
  }
}

// ---------------------------------------------------------------------------
// Kernel 3: merge column-segment partials per row -> loss, then mean.
// ---------------------------------------------------------------------------
__global__ void __launch_bounds__(256)
supcon_reduce(const float* __restrict__ pm, const float* __restrict__ pp,
              const float* __restrict__ pn, float* __restrict__ out) {
  __shared__ float sm[256];
  float s = 0.0f;
  for (int i = threadIdx.x; i < NTOT; i += 256) {
    float m0 = pm[i], m1 = pm[NTOT + i];
    float nm = fmaxf(m0, m1);
    float s0 = EXP2F(m0 - nm), s1 = EXP2F(m1 - nm);
    float P = pp[i] * s0 + pp[NTOT + i] * s1;
    float N = pn[i] * s0 + pn[NTOT + i] * s1;
    // loss_i = -10 * ln(0.5 * P / N)
    float loss = -LOSS_SCALE * LN2 * (LOG2F(P) - LOG2F(N) + LOG2_LAMDA);
    s += loss;
  }
  sm[threadIdx.x] = s;
  __syncthreads();
  for (int w = 128; w > 0; w >>= 1) {
    if (threadIdx.x < w) sm[threadIdx.x] += sm[threadIdx.x + w];
    __syncthreads();
  }
  if (threadIdx.x == 0) out[0] = sm[0] * (1.0f / NTOT);
}

extern "C" void kernel_launch(void* const* d_in, const int* in_sizes, int n_in,
                              void* d_out, int out_size, void* d_ws, size_t ws_size,
                              hipStream_t stream) {
  (void)in_sizes; (void)n_in; (void)out_size; (void)ws_size;
  const float* feat   = (const float*)d_in[0];   // [4096, 2, 128] fp32
  const int*   labels = (const int*)d_in[1];     // [4096]

  // ws layout: cf_hi (2MB) | cf_lo (2MB) | lab (32KB) | pm|pp|pn (3*64KB)
  char* ws = (char*)d_ws;
  _Float16* cf_hi = (_Float16*)ws;
  _Float16* cf_lo = (_Float16*)(ws + (size_t)NTOT * DIM * 2);
  int*      lab   = (int*)(ws + (size_t)NTOT * DIM * 4);
  float*    pm    = (float*)(ws + (size_t)NTOT * DIM * 4 + NTOT * 4);
  float*    pp    = pm + NSEG * NTOT;
  float*    pn    = pp + NSEG * NTOT;

  supcon_pack<<<(NTOT * DIM + 255) / 256, 256, 0, stream>>>(feat, labels, cf_hi, cf_lo, lab);
  supcon_main<<<(NTOT / 16) * NSEG, 32, 0, stream>>>(cf_hi, cf_lo, lab, pm, pp, pn);
  supcon_reduce<<<1, 256, 0, stream>>>(pm, pp, pn, (float*)d_out);
}
